// LightweightSTG_87385404604942
// MI455X (gfx1250) — matrix operand order repaired and checked by reading, not compiled, not run on hardware
//
#include <hip/hip_runtime.h>

// ---------------------------------------------------------------------------
// Fused  scores = sigmoid(x @ W^T + b)  ->  top-K mask  ->  (mask*x, scores)
// GEMM path: bf16x3 split-precision WMMA (v_wmma_f32_16x16x32_bf16), wave32.
// ---------------------------------------------------------------------------

typedef __bf16 v4bf  __attribute__((ext_vector_type(4)));
typedef __bf16 v8bf  __attribute__((ext_vector_type(8)));
typedef __bf16 v16bf __attribute__((ext_vector_type(16)));
typedef float  v8f   __attribute__((ext_vector_type(8)));

#define DIMD    1024   // feature dim (K of GEMM and N of GEMM)
#define MBLK    32     // rows per workgroup
#define NCHUNK  256    // feature columns per accumulation chunk
#define KCHUNK  64     // K tile (two WMMA k-steps of 32)
#define TSTRIDE 80     // bf16 LDS tile row stride (16B aligned, spreads banks)
#define LSTRIDE 1028   // f32 logits LDS row stride (padded: +4)
#define TOPK    307

// dynamic LDS layout (bytes)
#define OFF_LOGITS 0
#define SZ_LOGITS  (MBLK * LSTRIDE * 4)      // 131584
#define OFF_WHI    (OFF_LOGITS + SZ_LOGITS)
#define SZ_WT      (NCHUNK * TSTRIDE * 2)    // 40960
#define OFF_WLO    (OFF_WHI + SZ_WT)
#define OFF_XHI    (OFF_WLO + SZ_WT)
#define SZ_XT      (MBLK * TSTRIDE * 2)      // 5120
#define OFF_XLO    (OFF_XHI + SZ_XT)
#define OFF_BL     (OFF_XLO + SZ_XT)
#define SZ_BL      (DIMD * 4)
#define OFF_HIST   (OFF_BL + SZ_BL)
#define SZ_HIST    (8 * 256 * 4)
#define SMEM_TOTAL (OFF_HIST + SZ_HIST)      // 236032 B < 320 KB/WG

#define WMMA_BF16(A, B, C) \
  __builtin_amdgcn_wmma_f32_16x16x32_bf16(false, (A), false, (B), (short)0, (C), false, false)

__device__ __forceinline__ void cvt4(const float4 f, v4bf &h, v4bf &l) {
  __bf16 h0 = (__bf16)f.x, h1 = (__bf16)f.y, h2 = (__bf16)f.z, h3 = (__bf16)f.w;
  h[0] = h0; h[1] = h1; h[2] = h2; h[3] = h3;
  l[0] = (__bf16)(f.x - (float)h0);
  l[1] = (__bf16)(f.y - (float)h1);
  l[2] = (__bf16)(f.z - (float)h2);
  l[3] = (__bf16)(f.w - (float)h3);
}

// A fragment 16x32 bf16: lane holds M=lane&15; K runs {half*8..+7} and {16+half*8..+7}
__device__ __forceinline__ v16bf ldA(const __bf16 *base, int row, int s, int half) {
  const __bf16 *p = base + row * TSTRIDE + s * 32 + half * 8;
  union { v16bf v; v8bf h[2]; } u;
  u.h[0] = *(const v8bf *)p;
  u.h[1] = *(const v8bf *)(p + 16);
  return u.v;
}

// B fragment 32x16 bf16: lane holds N=lane&15; K runs {half*8..+7} and {16+half*8..+7}
__device__ __forceinline__ v16bf ldB(const __bf16 *base, int row, int s, int half) {
  const __bf16 *p = base + row * TSTRIDE + s * 32 + half * 8;
  union { v16bf v; v8bf h[2]; } u;
  u.h[0] = *(const v8bf *)p;
  u.h[1] = *(const v8bf *)(p + 16);
  return u.v;
}

__device__ __forceinline__ float sigmoidf(float z) {
  return 1.0f / (1.0f + __expf(-z));
}

__global__ __launch_bounds__(256) void stg_topk_fused(
    const float *__restrict__ X, const float *__restrict__ Wm,
    const float *__restrict__ Bv, float *__restrict__ out0,
    float *__restrict__ out1) {
  extern __shared__ char smem[];
  float    *logits = (float *)(smem + OFF_LOGITS);
  __bf16   *whi    = (__bf16 *)(smem + OFF_WHI);
  __bf16   *wlo    = (__bf16 *)(smem + OFF_WLO);
  __bf16   *xhi    = (__bf16 *)(smem + OFF_XHI);
  __bf16   *xlo    = (__bf16 *)(smem + OFF_XLO);
  float    *bl     = (float *)(smem + OFF_BL);
  unsigned *hist   = (unsigned *)(smem + OFF_HIST);

  const int tid  = threadIdx.x;
  const int lane = tid & 31;
  const int wv   = tid >> 5;
  const long long rowBlock = (long long)blockIdx.x * MBLK;

  // bias -> LDS (visible after first barrier)
  *(float4 *)&bl[tid * 4] = *(const float4 *)&Bv[tid * 4];

  const int tr = lane & 15;
  const int half = lane >> 4;

  v4bf rwh[16], rwl[16], rxh[2], rxl[2];

#pragma unroll 1
  for (int nc = 0; nc < DIMD / NCHUNK; ++nc) {
    v8f a00 = {}, a01 = {}, a10 = {}, a11 = {};

    // prime first k-tile into registers (global reads only)
    {
      const int k0 = 0;
#pragma unroll
      for (int i = 0; i < 16; ++i) {
        int q = tid + 256 * i; int r = q >> 4; int c = q & 15;
        float4 f = *(const float4 *)&Wm[(long long)(nc * NCHUNK + r) * DIMD + k0 + c * 4];
        cvt4(f, rwh[i], rwl[i]);
      }
#pragma unroll
      for (int i = 0; i < 2; ++i) {
        int q = tid + 256 * i; int r = q >> 4; int c = q & 15;
        float4 f = *(const float4 *)&X[(rowBlock + r) * DIMD + k0 + c * 4];
        cvt4(f, rxh[i], rxl[i]);
      }
    }

#pragma unroll 1
    for (int kt = 0; kt < DIMD / KCHUNK; ++kt) {
      __syncthreads();  // previous consumers of tile LDS are done
      // regs -> LDS (bf16 hi/lo tiles)
#pragma unroll
      for (int i = 0; i < 16; ++i) {
        int q = tid + 256 * i; int r = q >> 4; int c = q & 15;
        *(v4bf *)&whi[r * TSTRIDE + c * 4] = rwh[i];
        *(v4bf *)&wlo[r * TSTRIDE + c * 4] = rwl[i];
      }
#pragma unroll
      for (int i = 0; i < 2; ++i) {
        int q = tid + 256 * i; int r = q >> 4; int c = q & 15;
        *(v4bf *)&xhi[r * TSTRIDE + c * 4] = rxh[i];
        *(v4bf *)&xlo[r * TSTRIDE + c * 4] = rxl[i];
      }
      __syncthreads();  // tiles visible

      // software pipeline: issue next tile's global loads before the WMMAs
      if (kt + 1 < DIMD / KCHUNK) {
        const int k0 = (kt + 1) * KCHUNK;
#pragma unroll
        for (int i = 0; i < 16; ++i) {
          int q = tid + 256 * i; int r = q >> 4; int c = q & 15;
          float4 f = *(const float4 *)&Wm[(long long)(nc * NCHUNK + r) * DIMD + k0 + c * 4];
          cvt4(f, rwh[i], rwl[i]);
        }
#pragma unroll
        for (int i = 0; i < 2; ++i) {
          int q = tid + 256 * i; int r = q >> 4; int c = q & 15;
          float4 f = *(const float4 *)&X[(rowBlock + r) * DIMD + k0 + c * 4];
          cvt4(f, rxh[i], rxl[i]);
        }
      }

      // wave w: m-tiles {0,1}, n-tiles {2w, 2w+1}; bf16x3: hh + hl + lh
#pragma unroll
      for (int s = 0; s < 2; ++s) {
        v16bf ah0 = ldA(xhi, tr, s, half);
        v16bf ah1 = ldA(xhi, 16 + tr, s, half);
        v16bf al0 = ldA(xlo, tr, s, half);
        v16bf al1 = ldA(xlo, 16 + tr, s, half);
#pragma unroll
        for (int ni = 0; ni < 2; ++ni) {
          int wr = (2 * wv + ni) * 16 + tr;
          v16bf bh = ldB(whi, wr, s, half);
          v16bf bo = ldB(wlo, wr, s, half);
          if (ni == 0) {
            a00 = WMMA_BF16(ah0, bh, a00);
            a00 = WMMA_BF16(ah0, bo, a00);
            a00 = WMMA_BF16(al0, bh, a00);
            a10 = WMMA_BF16(ah1, bh, a10);
            a10 = WMMA_BF16(ah1, bo, a10);
            a10 = WMMA_BF16(al1, bh, a10);
          } else {
            a01 = WMMA_BF16(ah0, bh, a01);
            a01 = WMMA_BF16(ah0, bo, a01);
            a01 = WMMA_BF16(al0, bh, a01);
            a11 = WMMA_BF16(ah1, bh, a11);
            a11 = WMMA_BF16(ah1, bo, a11);
            a11 = WMMA_BF16(al1, bh, a11);
          }
        }
      }
    }

    // epilogue: acc + bias -> logits LDS (C layout: row = half*8+v, col = lane&15)
#pragma unroll
    for (int ni = 0; ni < 2; ++ni) {
      int gcol = nc * NCHUNK + (2 * wv + ni) * 16 + tr;
      float bb = bl[gcol];
      v8f c0 = ni ? a01 : a00;
      v8f c1 = ni ? a11 : a10;
#pragma unroll
      for (int v = 0; v < 8; ++v) {
        int r0 = half * 8 + v;
        logits[r0 * LSTRIDE + gcol] = c0[v] + bb;
        logits[(16 + r0) * LSTRIDE + gcol] = c1[v] + bb;
      }
    }
  }

  __syncthreads();  // all logits visible

  // ---- exact per-row top-K via 4-pass MSB radix select (1 wave per row) ----
  unsigned *myhist = hist + wv * 256;
#pragma unroll 1
  for (int rr = 0; rr < 4; ++rr) {
    const int r = wv * 4 + rr;
    const float *lrow = logits + r * LSTRIDE;

    unsigned prefix = 0;
    int remK = TOPK;
#pragma unroll 1
    for (int pass = 0; pass < 4; ++pass) {
      const int shift = 24 - 8 * pass;
      const unsigned pm = pass ? (0xFFFFFFFFu << (shift + 8)) : 0u;
#pragma unroll
      for (int t = 0; t < 8; ++t) myhist[lane * 8 + t] = 0u;
      asm volatile("" ::: "memory");
#pragma unroll 1
      for (int j = 0; j < 32; ++j) {
        unsigned bits = __float_as_uint(lrow[j * 32 + lane]);
        unsigned key = (bits & 0x80000000u) ? ~bits : (bits | 0x80000000u);
        if ((key & pm) == prefix) atomicAdd(&myhist[(key >> shift) & 255u], 1u);
      }
      asm volatile("s_wait_dscnt 0" ::: "memory");  // LDS atomics retired
      unsigned h8[8]; unsigned c = 0;
#pragma unroll
      for (int t = 0; t < 8; ++t) { h8[t] = myhist[lane * 8 + t]; c += h8[t]; }
      // wave suffix sum: count of keys in bucket-groups strictly above mine
      unsigned ts = c;
      for (int d = 1; d < 32; d <<= 1) {
        unsigned o = __shfl_down(ts, d);
        if (lane + d < 32) ts += o;
      }
      unsigned cum = ts - c;
      int fb = -1, nrem = 0;
#pragma unroll
      for (int b = 7; b >= 0; --b) {
        if (fb < 0) {
          if ((unsigned)remK > cum && (unsigned)remK <= cum + h8[b]) {
            fb = b; nrem = remK - (int)cum;
          }
          cum += h8[b];
        }
      }
      unsigned long long bal = __ballot(fb >= 0);
      int src = __ffsll(bal) - 1;
      int bloc = lane * 8 + fb;
      int Bsel = __shfl(bloc, src);
      remK = __shfl(nrem, src);
      prefix |= ((unsigned)Bsel) << shift;
    }
    const unsigned T = prefix;  // key of K-th largest
    const int R = remK;         // how many ties at T to select (lowest index first)

    // build selection mask; ballot word j covers cols j*32..j*32+31
    unsigned myword = 0;
    int tiesBefore = 0;
#pragma unroll 1
    for (int j = 0; j < 32; ++j) {
      unsigned bits = __float_as_uint(lrow[j * 32 + lane]);
      unsigned key = (bits & 0x80000000u) ? ~bits : (bits | 0x80000000u);
      bool eq = (key == T);
      unsigned ebal = (unsigned)__ballot(eq);
      int rank = tiesBefore + __popc(ebal & ((1u << lane) - 1u));
      bool sel = (key > T) || (eq && rank < R);
      unsigned w32 = (unsigned)__ballot(sel);
      tiesBefore += __popc(ebal);
      if (lane == j) myword = w32;
    }

    // coalesced float4 writes: masked x and sigmoid scores
    const long long grow = rowBlock + r;
    const float4 *xr = (const float4 *)(X + grow * DIMD);
    float4 *o0 = (float4 *)(out0 + grow * DIMD);
    float4 *o1 = (float4 *)(out1 + grow * DIMD);
#pragma unroll
    for (int i = 0; i < 8; ++i) {
      int c4 = lane + 32 * i;  // float4 index; cols c4*4..c4*4+3
      float4 xv = xr[c4];
      unsigned wbits = (unsigned)__shfl((int)myword, (lane >> 3) + 4 * i);
      unsigned m4 = (wbits >> ((lane & 7) * 4)) & 0xFu;
      float4 ov;
      ov.x = (m4 & 1u) ? xv.x : 0.0f;
      ov.y = (m4 & 2u) ? xv.y : 0.0f;
      ov.z = (m4 & 4u) ? xv.z : 0.0f;
      ov.w = (m4 & 8u) ? xv.w : 0.0f;
      o0[c4] = ov;
      float4 lg = *(const float4 *)&lrow[c4 * 4];
      float4 sv;
      sv.x = sigmoidf(lg.x);
      sv.y = sigmoidf(lg.y);
      sv.z = sigmoidf(lg.z);
      sv.w = sigmoidf(lg.w);
      o1[c4] = sv;
    }
  }
}

extern "C" void kernel_launch(void *const *d_in, const int *in_sizes, int n_in,
                              void *d_out, int out_size, void *d_ws,
                              size_t ws_size, hipStream_t stream) {
  const float *X  = (const float *)d_in[0];
  const float *Wm = (const float *)d_in[1];
  const float *Bv = (const float *)d_in[2];
  float *out0 = (float *)d_out;
  const long long n = (long long)in_sizes[0] / DIMD;  // 65536
  float *out1 = out0 + n * DIMD;
  const int grid = (int)(n / MBLK);  // 2048 workgroups
  stg_topk_fused<<<grid, 256, SMEM_TOTAL, stream>>>(X, Wm, Bv, out0, out1);
}
